// MultiheadAttention_36885179138314
// MI455X (gfx1250) — matrix operand order repaired and checked by reading, not compiled, not run on hardware
//
#include <hip/hip_runtime.h>
#include <hip/hip_bf16.h>

// ---------------------------------------------------------------------------
// CDNA5 (gfx1250, wave32) multi-head attention, bf16 WMMA with f32 accumulate.
//   proj GEMMs (f32 -> bf16) -> flash attention (bf16) -> out GEMM (bf16 -> f32)
// Round 3: softmax in exp2 domain with 0.125*log2(e) folded into the q
//          projection epilogue (removes 2 VALU muls per score element);
//          causal mask applied only on diagonal key tiles (uniform branch).
// ---------------------------------------------------------------------------

typedef __bf16 bf16_t;
typedef __attribute__((ext_vector_type(16))) bf16_t v16bf;
typedef __attribute__((ext_vector_type(8)))  float  v8f;

#define D_MODEL 1024
#define S_LEN   2048
#define BATCH   4
#define N_HEAD  16
#define HID     64
#define MASKED_C 1

union Frag16 { v16bf v; uint4 u[2]; };
union Pack4  { uint2 u; bf16_t h[4]; };
union Pack2  { unsigned u; bf16_t h[2]; };

static __device__ inline v16bf ld_frag_2x16B(const bf16_t* p0, const bf16_t* p1) {
    Frag16 f;
    f.u[0] = *(const uint4*)p0;
    f.u[1] = *(const uint4*)p1;
    return f.v;
}
static __device__ inline v16bf ld_frag_32B(const bf16_t* p) {
    Frag16 f;
    f.u[0] = ((const uint4*)p)[0];
    f.u[1] = ((const uint4*)p)[1];
    return f.v;
}
static __device__ inline v8f wmma_bf16(v16bf a, v16bf b, v8f c) {
    return __builtin_amdgcn_wmma_f32_16x16x32_bf16(false, a, false, b,
                                                   (short)0, c, false, false);
}
static __device__ inline v8f v8f_zero() {
    return (v8f){0.f, 0.f, 0.f, 0.f, 0.f, 0.f, 0.f, 0.f};
}
static __device__ inline float fast_exp2(float x) {
#if __has_builtin(__builtin_amdgcn_exp2f)
    return __builtin_amdgcn_exp2f(x);   // v_exp_f32 is natively 2^x
#else
    return exp2f(x);
#endif
}

// ---------------------------------------------------------------------------
// GEMM:  C[M,N] = (A[M,K] @ W[N,K]^T + bias) * oscale
// Block = 128x128 tile, 256 threads (8 waves), wave tile = 64x32, K step = 32.
// Global->register prefetch of tile k+32 overlaps the WMMA chain on tile k.
// ---------------------------------------------------------------------------
template <bool A_F32, bool OUT_F32>
__global__ __launch_bounds__(256)
void gemm_xwT_bias(const void* __restrict__ Aptr,
                   const float* __restrict__ W,
                   const float* __restrict__ bias,
                   void* __restrict__ Cptr,
                   int M, int N, int K, float oscale) {
    __shared__ __align__(16) bf16_t As[128][32];
    __shared__ __align__(16) bf16_t Bs[128][32];

    const int tid  = threadIdx.x;
    const int wave = tid >> 5;
    const int lane = tid & 31;
    const int half = lane >> 4;
    const int ln   = lane & 15;
    const int wm   = wave >> 2;   // 0..1  (64-row slab)
    const int wn   = wave & 3;    // 0..3  (32-col slab)
    const int m0   = blockIdx.y * 128;
    const int n0   = blockIdx.x * 128;

    const float*  Af = (const float*)Aptr;
    const bf16_t* Ab = (const bf16_t*)Aptr;

    v8f acc[4][2];
#pragma unroll
    for (int i = 0; i < 4; ++i)
#pragma unroll
        for (int j = 0; j < 2; ++j)
            acc[i][j] = v8f_zero();

    float4 aF[4];      // f32 A prefetch regs
    uint2  aB[4];      // bf16 A prefetch regs
    float4 wF[4];      // f32 W prefetch regs

    auto load_tile = [&](int k0) {
#pragma unroll
        for (int i = 0; i < 4; ++i) {
            const int c   = tid + i * 256;      // 0..1023 chunks of 4 elems
            const int row = c >> 3;
            const int col = (c & 7) * 4;
            if constexpr (A_F32)
                aF[i] = *(const float4*)(Af + (size_t)(m0 + row) * K + k0 + col);
            else
                aB[i] = *(const uint2*)(Ab + (size_t)(m0 + row) * K + k0 + col);
            wF[i] = *(const float4*)(W + (size_t)(n0 + row) * K + k0 + col);
        }
    };
    auto stage_tile = [&]() {
#pragma unroll
        for (int i = 0; i < 4; ++i) {
            const int c   = tid + i * 256;
            const int row = c >> 3;
            const int col = (c & 7) * 4;
            Pack4 pa;
            if constexpr (A_F32) {
                pa.h[0] = (bf16_t)aF[i].x; pa.h[1] = (bf16_t)aF[i].y;
                pa.h[2] = (bf16_t)aF[i].z; pa.h[3] = (bf16_t)aF[i].w;
            } else {
                pa.u = aB[i];
            }
            *(uint2*)&As[row][col] = pa.u;
            Pack4 pw;
            pw.h[0] = (bf16_t)wF[i].x; pw.h[1] = (bf16_t)wF[i].y;
            pw.h[2] = (bf16_t)wF[i].z; pw.h[3] = (bf16_t)wF[i].w;
            *(uint2*)&Bs[row][col] = pw.u;
        }
    };

    load_tile(0);
    for (int k0 = 0; k0 < K; k0 += 32) {
        stage_tile();
        __syncthreads();
        if (k0 + 32 < K) load_tile(k0 + 32);   // overlaps WMMA chain below

        v16bf bfr[2];
#pragma unroll
        for (int ns = 0; ns < 2; ++ns) {
            const int br = wn * 32 + ns * 16 + ln;      // B col = lane%16
            bfr[ns] = ld_frag_32B(&Bs[br][half * 16]);  // K = half*16 + e
        }
#pragma unroll
        for (int ms = 0; ms < 4; ++ms) {
            const int ar = wm * 64 + ms * 16 + ln;      // A row = lane%16
            const v16bf af = ld_frag_32B(&As[ar][half * 16]);
#pragma unroll
            for (int ns = 0; ns < 2; ++ns)
                acc[ms][ns] = wmma_bf16(af, bfr[ns], acc[ms][ns]);
        }
        __syncthreads();
    }

    // ---- epilogue: (acc + bias) * oscale ----
#pragma unroll
    for (int ms = 0; ms < 4; ++ms) {
#pragma unroll
        for (int ns = 0; ns < 2; ++ns) {
            const int n = n0 + wn * 32 + ns * 16 + ln;
            const float bv = bias[n];
#pragma unroll
            for (int r = 0; r < 8; ++r) {
                const int m = m0 + wm * 64 + ms * 16 + half * 8 + r;
                const float v = (acc[ms][ns][r] + bv) * oscale;
                if (OUT_F32) ((float*)Cptr)[(size_t)m * N + n] = v;
                else         ((bf16_t*)Cptr)[(size_t)m * N + n] = (bf16_t)v;
            }
        }
    }
}

// ---------------------------------------------------------------------------
// Flash attention, causal, exp2-domain online softmax. q rows are pre-scaled
// by 0.125*log2(e), so scores come out of the WMMA already in log2 units:
// softmax weight = 2^(s - m). Block = (batch, head, 128-row q tile); 8 waves
// x 16 q rows; 64-key streamed tiles.
// ---------------------------------------------------------------------------
__global__ __launch_bounds__(256)
void flash_attn_kernel(const bf16_t* __restrict__ Qb,
                       const bf16_t* __restrict__ Kb,
                       const bf16_t* __restrict__ Vb,
                       bf16_t* __restrict__ Ob) {
    __shared__ __align__(16) bf16_t Ks[64][64];     // [key][dim]
    __shared__ __align__(16) bf16_t Vt[64][64];     // [dim][key]
    __shared__ __align__(16) bf16_t Ps[8][16][64];  // per-wave P tile

    const int tid  = threadIdx.x;
    const int wave = tid >> 5;
    const int lane = tid & 31;
    const int half = lane >> 4;
    const int ln   = lane & 15;

    const int qt = blockIdx.x;             // q tile (0..15)
    const int h  = blockIdx.y;
    const int b  = blockIdx.z;
    const int q0 = qt * 128;
    const size_t baseRow = (size_t)b * S_LEN;
    const int hoff = h * HID;

    // Q fragments: wave owns rows q0 + wave*16 + (lane%16), K(dim) = 0..63
    const bf16_t* qp = Qb + (baseRow + q0 + wave * 16 + ln) * D_MODEL + hoff;
    v16bf qf[2];
#pragma unroll
    for (int f = 0; f < 2; ++f)
        qf[f] = ld_frag_32B(qp + f * 32 + half * 16);

    float mrun[8], lrun[8];
    v8f oacc[4];
#pragma unroll
    for (int r = 0; r < 8; ++r) { mrun[r] = -3.0e38f; lrun[r] = 0.f; }
#pragma unroll
    for (int t = 0; t < 4; ++t) oacc[t] = v8f_zero();

    // ---- K/V global->register prefetch state ----
    uint2 gK[4];            // K tile: 64x64, 4 chunks/thread
    uint2 gVa[2], gVb[2];   // V tile: 2 chunks/thread, paired key rows

    auto prefetch_kv = [&](int k0) {
#pragma unroll
        for (int i = 0; i < 4; ++i) {
            const int c = tid + i * 256;            // 0..1023
            gK[i] = *(const uint2*)(Kb + (baseRow + k0 + (c >> 4)) * D_MODEL
                                    + hoff + (c & 15) * 4);
        }
#pragma unroll
        for (int i = 0; i < 2; ++i) {
            const int c  = tid + i * 256;           // 0..511
            const int rp = (c >> 4) * 2;            // key row pair
            const int vc = (c & 15) * 4;            // dim chunk
            gVa[i] = *(const uint2*)(Vb + (baseRow + k0 + rp) * D_MODEL + hoff + vc);
            gVb[i] = *(const uint2*)(Vb + (baseRow + k0 + rp + 1) * D_MODEL + hoff + vc);
        }
    };
    auto stage_kv = [&]() {
#pragma unroll
        for (int i = 0; i < 4; ++i) {
            const int c = tid + i * 256;
            *(uint2*)&Ks[c >> 4][(c & 15) * 4] = gK[i];
        }
#pragma unroll
        for (int i = 0; i < 2; ++i) {
            const int c  = tid + i * 256;
            const int rp = (c >> 4) * 2;
            const int vc = (c & 15) * 4;
            Pack4 pa, pb;
            pa.u = gVa[i]; pb.u = gVb[i];
#pragma unroll
            for (int j = 0; j < 4; ++j) {           // transpose: packed b32 store
                Pack2 w;
                w.h[0] = pa.h[j]; w.h[1] = pb.h[j];
                *(unsigned*)&Vt[vc + j][rp] = w.u;
            }
        }
    };

    const int ktMax = MASKED_C ? (2 * qt + 1) : (S_LEN / 64 - 1);
    prefetch_kv(0);
    for (int kt = 0; kt <= ktMax; ++kt) {
        const int k0 = kt * 64;
        stage_kv();
        __syncthreads();
        if (kt < ktMax) prefetch_kv(k0 + 64);       // overlaps all compute below

        // ---- S = Q @ K^T : 4 key subtiles x 2 dim steps (log2 units) ----
        v8f sc[4];
#pragma unroll
        for (int t = 0; t < 4; ++t) {
            v8f a = v8f_zero();
            const int kn = t * 16 + ln;              // key index = B col
#pragma unroll
            for (int f = 0; f < 2; ++f)
                a = wmma_bf16(qf[f], ld_frag_32B(&Ks[kn][f * 32 + half * 16]), a);
            sc[t] = a;
        }

        // ---- causal mask: only the two diagonal key tiles can intersect ----
        const int mg0 = q0 + wave * 16 + half * 8;
        if (MASKED_C && kt >= 2 * qt) {              // block-uniform branch
#pragma unroll
            for (int t = 0; t < 4; ++t) {
                const int ng = k0 + t * 16 + ln;
#pragma unroll
                for (int r = 0; r < 8; ++r)
                    if (ng > (mg0 + r)) sc[t][r] = -3.0e38f;
            }
        }

        // ---- exp2-domain online softmax (rows replicated over 16 lanes) ----
        float rowm[8];
#pragma unroll
        for (int r = 0; r < 8; ++r) rowm[r] = -3.0e38f;
#pragma unroll
        for (int t = 0; t < 4; ++t)
#pragma unroll
            for (int r = 0; r < 8; ++r)
                rowm[r] = fmaxf(rowm[r], sc[t][r]);
#pragma unroll
        for (int r = 0; r < 8; ++r) {
            rowm[r] = fmaxf(rowm[r], __shfl_xor(rowm[r], 8));
            rowm[r] = fmaxf(rowm[r], __shfl_xor(rowm[r], 4));
            rowm[r] = fmaxf(rowm[r], __shfl_xor(rowm[r], 2));
            rowm[r] = fmaxf(rowm[r], __shfl_xor(rowm[r], 1));
        }
        float nm[8], scl[8], rsum[8];
#pragma unroll
        for (int r = 0; r < 8; ++r) {
            nm[r]   = fmaxf(mrun[r], rowm[r]);
            scl[r]  = fast_exp2(mrun[r] - nm[r]);
            rsum[r] = 0.f;
        }
#pragma unroll
        for (int t = 0; t < 4; ++t)
#pragma unroll
            for (int r = 0; r < 8; ++r) {
                const float p = fast_exp2(sc[t][r] - nm[r]);
                sc[t][r] = p;
                rsum[r] += p;
            }
#pragma unroll
        for (int r = 0; r < 8; ++r) {
            rsum[r] += __shfl_xor(rsum[r], 8);
            rsum[r] += __shfl_xor(rsum[r], 4);
            rsum[r] += __shfl_xor(rsum[r], 2);
            rsum[r] += __shfl_xor(rsum[r], 1);
            lrun[r] = lrun[r] * scl[r] + rsum[r];
            mrun[r] = nm[r];
        }
#pragma unroll
        for (int t = 0; t < 4; ++t)
#pragma unroll
            for (int r = 0; r < 8; ++r)
                oacc[t][r] *= scl[r];

        // ---- P: C layout -> wave-private LDS tile ----
#pragma unroll
        for (int t = 0; t < 4; ++t)
#pragma unroll
            for (int r = 0; r < 8; ++r)
                Ps[wave][half * 8 + r][t * 16 + ln] = (bf16_t)sc[t][r];
        __builtin_amdgcn_wave_barrier();   // wave-private: order DS store->load

        // ---- O += P @ V : 4 dim subtiles x 2 key steps ----
#pragma unroll
        for (int t2 = 0; t2 < 4; ++t2) {
            v8f a = oacc[t2];
#pragma unroll
            for (int kf = 0; kf < 2; ++kf) {
                const v16bf pa = ld_frag_32B(&Ps[wave][ln][kf * 32 + half * 16]);
                const v16bf vb = ld_frag_32B(&Vt[t2 * 16 + ln][kf * 32 + half * 16]);
                a = wmma_bf16(pa, vb, a);
            }
            oacc[t2] = a;
        }
        __syncthreads();   // protect Ks/Vt before next stage
    }

    // ---- normalize + store bf16 attention output ----
    const size_t orow = baseRow + q0 + wave * 16 + half * 8;
#pragma unroll
    for (int t2 = 0; t2 < 4; ++t2)
#pragma unroll
        for (int r = 0; r < 8; ++r) {
            const float v = oacc[t2][r] / lrun[r];
            Ob[(orow + r) * D_MODEL + hoff + t2 * 16 + ln] = (bf16_t)v;
        }
}

// ---------------------------------------------------------------------------
// Launch: Q/K/V proj GEMMs -> flash attention -> output proj GEMM.
// ws layout: qb | kb | vb | attn  (each 8192*1024 bf16 = 16 MB)
// ---------------------------------------------------------------------------
extern "C" void kernel_launch(void* const* d_in, const int* in_sizes, int n_in,
                              void* d_out, int out_size, void* d_ws, size_t ws_size,
                              hipStream_t stream) {
    (void)in_sizes; (void)n_in; (void)out_size; (void)ws_size;
    const float* Q  = (const float*)d_in[0];
    const float* K  = (const float*)d_in[1];
    const float* V  = (const float*)d_in[2];
    const float* Wq = (const float*)d_in[3];
    const float* bq = (const float*)d_in[4];
    const float* Wk = (const float*)d_in[5];
    const float* bk = (const float*)d_in[6];
    const float* Wv = (const float*)d_in[7];
    const float* bv = (const float*)d_in[8];
    const float* Wo = (const float*)d_in[9];
    const float* bo = (const float*)d_in[10];

    const int M = BATCH * S_LEN;   // 8192
    const int D = D_MODEL;         // 1024
    const size_t bufB = (size_t)M * D * sizeof(bf16_t);   // 16 MB

    char* ws = (char*)d_ws;
    void* qb = (void*)(ws + 0 * bufB);
    void* kb = (void*)(ws + 1 * bufB);
    void* vb = (void*)(ws + 2 * bufB);
    void* ab = (void*)(ws + 3 * bufB);

    dim3 gg(D / 128, M / 128);     // (8, 64)
    dim3 blk(256);

    // q pre-scaled by (1/sqrt(HID)) * log2(e): scores leave WMMA in log2 units
    const float qscale = 0.125f * 1.44269504088896340736f;

    gemm_xwT_bias<true, false><<<gg, blk, 0, stream>>>(Q, Wq, bq, qb, M, D, D, qscale);
    gemm_xwT_bias<true, false><<<gg, blk, 0, stream>>>(K, Wk, bk, kb, M, D, D, 1.0f);
    gemm_xwT_bias<true, false><<<gg, blk, 0, stream>>>(V, Wv, bv, vb, M, D, D, 1.0f);

    dim3 ga(S_LEN / 128, N_HEAD, BATCH);   // (16, 16, 4)
    flash_attn_kernel<<<ga, blk, 0, stream>>>((const bf16_t*)qb, (const bf16_t*)kb,
                                              (const bf16_t*)vb, (bf16_t*)ab);

    gemm_xwT_bias<false, true><<<gg, blk, 0, stream>>>(ab, Wo, bo, d_out, M, D, D, 1.0f);
}
